// MobileMambaHART_57432302682148
// MI455X (gfx1250) — matrix operand order repaired and checked by reading, not compiled, not run on hardware
//
#include <hip/hip_runtime.h>
#include <hip/hip_bf16.h>

typedef __attribute__((ext_vector_type(16))) _Float16 v16h;
typedef __attribute__((ext_vector_type(8)))  _Float16 v8h;
typedef __attribute__((ext_vector_type(4)))  _Float16 v4h;
typedef __attribute__((ext_vector_type(8)))  float    v8f;

#define B_    64
#define T_    16384
#define D_    128
#define E_    256
#define N_    8
#define S_    1024
#define NCLS_ 18
#define M_    (B_ * S_)   // 65536 rows
#define EPS_  1e-5f

// Use gfx1250 async global->LDS DMA (ASYNCcnt path) for weight staging.
#define TRY_ASYNC_LDS 1

__device__ __forceinline__ float fast_silu(float v) {
    // v * sigmoid(v); v_rcp_f32 instead of full IEEE divide
    return v * __builtin_amdgcn_rcpf(1.f + __expf(-v));
}

// Stage `halfs` fp16 elements from global into LDS with all 256 threads.
__device__ __forceinline__ void stage_weights(_Float16* lds, const _Float16* __restrict__ g,
                                              int halfs) {
    int t = threadIdx.x;
#if TRY_ASYNC_LDS
    for (int i = t * 8; i < halfs; i += 256 * 8) {
        unsigned int       dst = (unsigned int)(size_t)(lds + i);   // low 32b = LDS offset
        unsigned long long src = (unsigned long long)(size_t)(g + i);
        asm volatile("global_load_async_to_lds_b128 %0, %1, off"
                     :: "v"(dst), "v"(src) : "memory");
    }
    asm volatile("s_wait_asynccnt 0x0" ::: "memory");
#else
    for (int i = t * 8; i < halfs; i += 256 * 8)
        *(v8h*)(lds + i) = *(const v8h*)(g + i);
#endif
    __syncthreads();
}

// ---------------------------------------------------------------------------
// fp32 -> fp16 weight conversion
// ---------------------------------------------------------------------------
__global__ void k_cvt_f16(const float* __restrict__ src, _Float16* __restrict__ dst, int n) {
    int i = blockIdx.x * 256 + threadIdx.x;
    if (i < n) dst[i] = (_Float16)src[i];
}

// ---------------------------------------------------------------------------
// Stem: dual conv1d (stride 16, k=16, VALID) + pos_emb -> x f32 [B*S, 128]
// ---------------------------------------------------------------------------
__global__ void k_stem(const float* __restrict__ inp,    // [B,T,6]
                       const float* __restrict__ accw,   // [64,3,16]
                       const float* __restrict__ accb,   // [64]
                       const float* __restrict__ gyrow,  // [64,3,16]
                       const float* __restrict__ gyrob,  // [64]
                       const float* __restrict__ pos,    // [S,128]
                       float* __restrict__ x)            // [B*S,128]
{
    int bs = blockIdx.x;
    int b  = bs >> 10;
    int s  = bs & (S_ - 1);
    __shared__ float tile[16 * 6];
    int t = threadIdx.x;
    if (t < 96) tile[t] = inp[((size_t)b * T_ + (size_t)s * 16) * 6 + t];
    __syncthreads();
    int d = t;
    float sum;
    if (d < 64) {
        const float* w = accw + d * 48;
        sum = accb[d];
        #pragma unroll
        for (int c = 0; c < 3; ++c)
            #pragma unroll
            for (int p = 0; p < 16; ++p)
                sum += w[c * 16 + p] * tile[p * 6 + c];
    } else {
        const float* w = gyrow + (d - 64) * 48;
        sum = gyrob[d - 64];
        #pragma unroll
        for (int c = 0; c < 3; ++c)
            #pragma unroll
            for (int p = 0; p < 16; ++p)
                sum += w[c * 16 + p] * tile[p * 6 + (c + 3)];
    }
    x[(size_t)bs * D_ + d] = sum + pos[s * D_ + d];
}

// ---------------------------------------------------------------------------
// LayerNorm (rows of 128) fp32 -> fp16 ; one wave32 per row, 4 elems/lane
// ---------------------------------------------------------------------------
__global__ void k_ln_f16(const float* __restrict__ x, const float* __restrict__ g,
                         const float* __restrict__ bb, _Float16* __restrict__ out)
{
    size_t row = (size_t)blockIdx.x * (blockDim.x >> 5) + (threadIdx.x >> 5);
    int lane = threadIdx.x & 31;
    const float* xr = x + row * D_;
    float4 v = *(const float4*)(xr + lane * 4);
    float s = v.x + v.y + v.z + v.w;
    #pragma unroll
    for (int o = 16; o; o >>= 1) s += __shfl_xor(s, o, 32);
    float mean = s * (1.f / 128.f);
    float dx = v.x - mean, dy = v.y - mean, dz = v.z - mean, dw = v.w - mean;
    float sv = dx * dx + dy * dy + dz * dz + dw * dw;
    #pragma unroll
    for (int o = 16; o; o >>= 1) sv += __shfl_xor(sv, o, 32);
    float inv = rsqrtf(sv * (1.f / 128.f) + EPS_);
    float4 gv = *(const float4*)(g + lane * 4);
    float4 bv = *(const float4*)(bb + lane * 4);
    v4h o4;
    o4[0] = (_Float16)(dx * inv * gv.x + bv.x);
    o4[1] = (_Float16)(dy * inv * gv.y + bv.y);
    o4[2] = (_Float16)(dz * inv * gv.z + bv.z);
    o4[3] = (_Float16)(dw * inv * gv.w + bv.w);
    *(v4h*)(out + row * D_ + lane * 4) = o4;
}

// ---------------------------------------------------------------------------
// WMMA GEMM #1: out = silu(A @ W^T + bias), A f16 [M,128], W f16 [256,128]
// Block: 8 waves, W staged in LDS (64 KB, async DMA). Each wave owns one
// 16-row M tile and 8 accumulator tiles; 1 global A fragment feeds 8 WMMAs.
// ---------------------------------------------------------------------------
__global__ void __launch_bounds__(256) k_gemm_pc1(const _Float16* __restrict__ A,
                                                  const _Float16* __restrict__ W,
                                                  const float* __restrict__ bias,
                                                  _Float16* __restrict__ out)
{
    __shared__ _Float16 wlds[E_ * D_];          // 256x128 f16 = 64 KB
    stage_weights(wlds, W, E_ * D_);

    int t = threadIdx.x;
    int wv = t >> 5, lane = t & 31;
    int mt = blockIdx.x * 8 + wv;               // M tile
    int r = lane & 15, hf = lane >> 4;
    const _Float16* arow = A + (size_t)(mt * 16 + r) * D_;

    #pragma unroll 1
    for (int nc = 0; nc < 16; nc += 8) {        // two passes of 8 N tiles
        v8f acc[8];
        v8f z = {};
        #pragma unroll
        for (int j = 0; j < 8; ++j) acc[j] = z;
        #pragma unroll
        for (int k0 = 0; k0 < D_; k0 += 32) {
            v8h alo = *(const v8h*)(arow + k0 + hf * 16);
            v8h ahi = *(const v8h*)(arow + k0 + hf * 16 + 8);
            v16h a;
            #pragma unroll
            for (int i = 0; i < 8; ++i) { a[i] = alo[i]; a[8 + i] = ahi[i]; }
            #pragma unroll
            for (int j = 0; j < 8; ++j) {
                v16h b = *(const v16h*)(wlds + (size_t)((nc + j) * 16 + r) * D_ + k0 + hf * 16);
                acc[j] = __builtin_amdgcn_wmma_f32_16x16x32_f16(false, a, false, b,
                                                                (short)0, acc[j], false, false);
            }
        }
        #pragma unroll
        for (int j = 0; j < 8; ++j) {
            int n = (nc + j) * 16 + r;
            float bn = bias[n];
            _Float16* op = out + (size_t)(mt * 16 + hf * 8) * E_ + n;
            #pragma unroll
            for (int i = 0; i < 8; ++i)
                op[(size_t)i * E_] = (_Float16)fast_silu(acc[j][i] + bn);
        }
    }
}

// ---------------------------------------------------------------------------
// Fused depthwise conv(k=5,pad=2) + BatchNorm + SiLU + SSM scan, in-place f16.
// One thread per (b,e); in-place safe (only reads xc[s+3] after writing y[s]).
// ---------------------------------------------------------------------------
__global__ void k_dwscan(_Float16* __restrict__ xc,          // [B*S, 256] in/out
                         const float* __restrict__ dww, const float* __restrict__ dwb,
                         const float* __restrict__ bng, const float* __restrict__ bnb,
                         const float* __restrict__ bnm, const float* __restrict__ bnv,
                         const float* __restrict__ Ap, const float* __restrict__ Bp,
                         const float* __restrict__ Cp, const float* __restrict__ Dp)
{
    int tid = blockIdx.x * blockDim.x + threadIdx.x;   // B*E total
    int b = tid >> 8;
    int e = tid & (E_ - 1);
    float w0 = dww[e * 5 + 0], w1 = dww[e * 5 + 1], w2 = dww[e * 5 + 2];
    float w3 = dww[e * 5 + 3], w4 = dww[e * 5 + 4];
    float scale = bng[e] * rsqrtf(bnv[e] + EPS_);
    float shift = (dwb[e] - bnm[e]) * scale + bnb[e];
    float Aexp = __expf(Ap[e]);
    float Bn[N_], Cn[N_], h[N_];
    #pragma unroll
    for (int n = 0; n < N_; ++n) { Bn[n] = Bp[e * N_ + n]; Cn[n] = Cp[e * N_ + n]; h[n] = 0.f; }
    float Dv = Dp[e];
    size_t base = (size_t)b * S_ * E_ + e;
    auto ld = [&](int s) -> float {
        return (s < S_) ? (float)xc[base + (size_t)s * E_] : 0.f;
    };
    float q0 = 0.f, q1 = 0.f, q2 = ld(0), q3 = ld(1), q4 = ld(2);
    for (int s = 0; s < S_; ++s) {
        float conv = w0 * q0 + w1 * q1 + w2 * q2 + w3 * q3 + w4 * q4;
        float u = fast_silu(conv * scale + shift);
        float y = Dv * u;
        #pragma unroll
        for (int n = 0; n < N_; ++n) { h[n] = Aexp * h[n] + u * Bn[n]; y += Cn[n] * h[n]; }
        xc[base + (size_t)s * E_] = (_Float16)y;
        q0 = q1; q1 = q2; q2 = q3; q3 = q4; q4 = ld(s + 3);
    }
}

// ---------------------------------------------------------------------------
// WMMA GEMM #2: x += A @ W^T + bias (residual, f32 out)
// A f16 [M,256], W f16 [128,256] staged in LDS (64 KB, async DMA).
// ---------------------------------------------------------------------------
__global__ void __launch_bounds__(256) k_gemm_pc2(const _Float16* __restrict__ A,
                                                  const _Float16* __restrict__ W,
                                                  const float* __restrict__ bias,
                                                  float* __restrict__ x)
{
    __shared__ _Float16 wlds[D_ * E_];          // 128x256 f16 = 64 KB
    stage_weights(wlds, W, D_ * E_);

    int t = threadIdx.x;
    int wv = t >> 5, lane = t & 31;
    int mt = blockIdx.x * 8 + wv;
    int r = lane & 15, hf = lane >> 4;
    const _Float16* arow = A + (size_t)(mt * 16 + r) * E_;

    v8f acc[8];
    v8f z = {};
    #pragma unroll
    for (int j = 0; j < 8; ++j) acc[j] = z;
    #pragma unroll
    for (int k0 = 0; k0 < E_; k0 += 32) {
        v8h alo = *(const v8h*)(arow + k0 + hf * 16);
        v8h ahi = *(const v8h*)(arow + k0 + hf * 16 + 8);
        v16h a;
        #pragma unroll
        for (int i = 0; i < 8; ++i) { a[i] = alo[i]; a[8 + i] = ahi[i]; }
        #pragma unroll
        for (int j = 0; j < 8; ++j) {
            v16h b = *(const v16h*)(wlds + (size_t)(j * 16 + r) * E_ + k0 + hf * 16);
            acc[j] = __builtin_amdgcn_wmma_f32_16x16x32_f16(false, a, false, b,
                                                            (short)0, acc[j], false, false);
        }
    }
    #pragma unroll
    for (int j = 0; j < 8; ++j) {
        int n = j * 16 + r;
        float bn = bias[n];
        float* op = x + (size_t)(mt * 16 + hf * 8) * D_ + n;
        #pragma unroll
        for (int i = 0; i < 8; ++i)
            op[(size_t)i * D_] = acc[j][i] + bn + op[(size_t)i * D_];
    }
}

// ---------------------------------------------------------------------------
// Final LayerNorm + mean over S -> pooled [B,128]; one block (8 waves) per b
// ---------------------------------------------------------------------------
__global__ void k_final_pool(const float* __restrict__ x, const float* __restrict__ g,
                             const float* __restrict__ bb, float* __restrict__ pooled)
{
    int b = blockIdx.x;
    int wv = threadIdx.x >> 5, lane = threadIdx.x & 31;
    float4 gv = *(const float4*)(g + lane * 4);
    float4 bv = *(const float4*)(bb + lane * 4);
    float a0 = 0.f, a1 = 0.f, a2 = 0.f, a3 = 0.f;
    for (int s = wv; s < S_; s += 8) {
        const float* xr = x + ((size_t)b * S_ + s) * D_;
        float4 v = *(const float4*)(xr + lane * 4);
        float sm = v.x + v.y + v.z + v.w;
        #pragma unroll
        for (int o = 16; o; o >>= 1) sm += __shfl_xor(sm, o, 32);
        float mean = sm * (1.f / 128.f);
        float dx = v.x - mean, dy = v.y - mean, dz = v.z - mean, dw = v.w - mean;
        float sv = dx * dx + dy * dy + dz * dz + dw * dw;
        #pragma unroll
        for (int o = 16; o; o >>= 1) sv += __shfl_xor(sv, o, 32);
        float inv = rsqrtf(sv * (1.f / 128.f) + EPS_);
        a0 += dx * inv * gv.x + bv.x;
        a1 += dy * inv * gv.y + bv.y;
        a2 += dz * inv * gv.z + bv.z;
        a3 += dw * inv * gv.w + bv.w;
    }
    __shared__ float red[8][128];
    red[wv][lane * 4 + 0] = a0;
    red[wv][lane * 4 + 1] = a1;
    red[wv][lane * 4 + 2] = a2;
    red[wv][lane * 4 + 3] = a3;
    __syncthreads();
    if (threadIdx.x < 128) {
        float s = 0.f;
        #pragma unroll
        for (int w = 0; w < 8; ++w) s += red[w][threadIdx.x];
        pooled[b * D_ + threadIdx.x] = s * (1.f / (float)S_);
    }
}

// ---------------------------------------------------------------------------
// Head: silu(pooled @ hw1 + hb1) @ hw2 + hb2 -> softmax ; one block per batch
// ---------------------------------------------------------------------------
__global__ void k_head(const float* __restrict__ pooled,
                       const float* __restrict__ hw1, const float* __restrict__ hb1,
                       const float* __restrict__ hw2, const float* __restrict__ hb2,
                       float* __restrict__ out)
{
    int b = blockIdx.x, t = threadIdx.x;    // 256 threads
    __shared__ float pd[128];
    __shared__ float h[256];
    __shared__ float logits[NCLS_];
    if (t < 128) pd[t] = pooled[b * D_ + t];
    __syncthreads();
    float s = hb1[t];
    for (int d = 0; d < 128; ++d) s += pd[d] * hw1[d * 256 + t];
    h[t] = fast_silu(s);
    __syncthreads();
    if (t < NCLS_) {
        float l = hb2[t];
        for (int j = 0; j < 256; ++j) l += h[j] * hw2[j * NCLS_ + t];
        logits[t] = l;
    }
    __syncthreads();
    if (t == 0) {
        float mx = -1e30f;
        for (int c = 0; c < NCLS_; ++c) mx = fmaxf(mx, logits[c]);
        float sum = 0.f;
        float ex[NCLS_];
        for (int c = 0; c < NCLS_; ++c) { ex[c] = __expf(logits[c] - mx); sum += ex[c]; }
        float isum = __builtin_amdgcn_rcpf(sum);
        for (int c = 0; c < NCLS_; ++c) out[b * NCLS_ + c] = ex[c] * isum;
    }
}

// ---------------------------------------------------------------------------
extern "C" void kernel_launch(void* const* d_in, const int* in_sizes, int n_in,
                              void* d_out, int out_size, void* d_ws, size_t ws_size,
                              hipStream_t stream)
{
    (void)in_sizes; (void)n_in; (void)out_size; (void)ws_size;
    const float* inputs  = (const float*)d_in[0];
    const float* acc_w   = (const float*)d_in[1];
    const float* acc_b   = (const float*)d_in[2];
    const float* gyro_w  = (const float*)d_in[3];
    const float* gyro_b  = (const float*)d_in[4];
    const float* pos_emb = (const float*)d_in[5];
    const float* ln_g    = (const float*)d_in[6];
    const float* ln_b    = (const float*)d_in[7];
    const float* pc1_w   = (const float*)d_in[8];
    const float* pc1_b   = (const float*)d_in[9];
    const float* dw_w    = (const float*)d_in[10];
    const float* dw_b    = (const float*)d_in[11];
    const float* bn_g    = (const float*)d_in[12];
    const float* bn_b    = (const float*)d_in[13];
    const float* bn_mean = (const float*)d_in[14];
    const float* bn_var  = (const float*)d_in[15];
    const float* Ap      = (const float*)d_in[16];
    const float* Bp      = (const float*)d_in[17];
    const float* Cp      = (const float*)d_in[18];
    const float* Dp      = (const float*)d_in[19];
    const float* pc2_w   = (const float*)d_in[20];
    const float* pc2_b   = (const float*)d_in[21];
    const float* fln_g   = (const float*)d_in[22];
    const float* fln_b   = (const float*)d_in[23];
    const float* hw1     = (const float*)d_in[24];
    const float* hb1     = (const float*)d_in[25];
    const float* hw2     = (const float*)d_in[26];
    const float* hb2     = (const float*)d_in[27];

    char* ws = (char*)d_ws;
    size_t off = 0;
    auto carve = [&](size_t bytes) -> void* {
        void* p = ws + off;
        off = (off + bytes + 255) & ~(size_t)255;
        return p;
    };
    float*     x      = (float*)    carve((size_t)M_ * D_ * 4);   // 33.5 MB
    _Float16*  xln    = (_Float16*) carve((size_t)M_ * D_ * 2);   // 16.8 MB
    _Float16*  xc     = (_Float16*) carve((size_t)M_ * E_ * 2);   // 33.5 MB (reused as scan out)
    _Float16*  pc1h   = (_Float16*) carve((size_t)3 * E_ * D_ * 2);
    _Float16*  pc2h   = (_Float16*) carve((size_t)3 * D_ * E_ * 2);
    float*     pooled = (float*)    carve((size_t)B_ * D_ * 4);

    {
        int n = 3 * E_ * D_;
        k_cvt_f16<<<(n + 255) / 256, 256, 0, stream>>>(pc1_w, pc1h, n);
        k_cvt_f16<<<(n + 255) / 256, 256, 0, stream>>>(pc2_w, pc2h, n);
    }

    k_stem<<<B_ * S_, 128, 0, stream>>>(inputs, acc_w, acc_b, gyro_w, gyro_b, pos_emb, x);

    for (int blk = 0; blk < 3; ++blk) {
        k_ln_f16<<<M_ / 8, 256, 0, stream>>>(x, ln_g + blk * D_, ln_b + blk * D_, xln);
        k_gemm_pc1<<<M_ / 16 / 8, 256, 0, stream>>>(
            xln, pc1h + (size_t)blk * E_ * D_, pc1_b + blk * E_, xc);
        k_dwscan<<<(B_ * E_) / 256, 256, 0, stream>>>(
            xc, dw_w + blk * E_ * 5, dw_b + blk * E_,
            bn_g + blk * E_, bn_b + blk * E_, bn_mean + blk * E_, bn_var + blk * E_,
            Ap + blk * E_, Bp + blk * E_ * N_, Cp + blk * E_ * N_, Dp + blk * E_);
        k_gemm_pc2<<<M_ / 16 / 8, 256, 0, stream>>>(
            xc, pc2h + (size_t)blk * D_ * E_, pc2_b + blk * D_, x);
    }

    k_final_pool<<<B_, 256, 0, stream>>>(x, fln_g, fln_b, pooled);
    k_head<<<B_, 256, 0, stream>>>(pooled, hw1, hb1, hw2, hb2, (float*)d_out);
}
